// _RHN_37520834298156
// MI455X (gfx1250) — matrix operand order repaired and checked by reading, not compiled, not run on hardware
//
#include <hip/hip_runtime.h>
#include <hip/hip_bf16.h>
#include <stdint.h>

#define S_LEN   512
#define B_DIM   64
#define IN_DIM  512
#define HID_DIM 512
#define DEPTH   5
#define NBLK    32
#define TPB     128   // 4 waves/block * 32 blocks = 128 waves = 128 output tiles
#define NGROUP  4     // independent recurrent chains, one per 16-row batch tile
#define NBLK_G  (NBLK / NGROUP)   // 8 blocks per sync group

typedef __attribute__((ext_vector_type(16))) __bf16        v16bf;
typedef __attribute__((ext_vector_type(8)))  float         v8f;
typedef __attribute__((ext_vector_type(8)))  unsigned short v8us;
typedef __attribute__((ext_vector_type(4)))  unsigned short v4us;

union FragU { v16bf bf; v8us h[2]; };

// A fragment (16x32 bf16, M x K): lane m=lane&15, half=lane>>4.
// elements 0..7  = A[m][8*half + 0..7], elements 8..15 = A[m][16 + 8*half + 0..7]
__device__ __forceinline__ v16bf ldA(const unsigned short* p) {
  FragU f;
  f.h[0] = *(const v8us*)(p);        // K = kbase + 8*half + 0..7
  f.h[1] = *(const v8us*)(p + 16);   // K = kbase + 16 + 8*half + 0..7
  return f.bf;
}
// B fragment (32x16 bf16, K x N): lane n=lane&15, half=lane>>4.
// elements 0..7 = B[8*half + 0..7][n], elements 8..15 = B[16 + 8*half + 0..7][n]  (stored as W[n][k], K-contiguous)
__device__ __forceinline__ v16bf ldB(const unsigned short* p) {
  FragU f;
  f.h[0] = *(const v8us*)(p);        // K = kbase + 8*half + 0..7
  f.h[1] = *(const v8us*)(p + 16);   // K = kbase + 16 + 8*half + 0..7
  return f.bf;
}

__device__ __forceinline__ v8f wmma_bf16(v16bf a, v16bf b, v8f c) {
  return __builtin_amdgcn_wmma_f32_16x16x32_bf16(false, a, false, b, (short)0, c,
                                                 false, false);
}

__device__ __forceinline__ unsigned short f2bf_rne(float x) {
  unsigned u = __float_as_uint(x);
  unsigned r = u + 0x7FFFu + ((u >> 16) & 1u);
  return (unsigned short)(r >> 16);
}

// Branchless activations on the native v_exp_f32 / v_rcp_f32 path (serial
// critical path: avoid the branchy ocml tanh with its EXEC save/restore).
__device__ __forceinline__ float sigmf(float x) {
  return __builtin_amdgcn_rcpf(1.0f + __expf(-x));
}
__device__ __forceinline__ float tanh_fast(float x) {
  float e = __expf(-2.0f * __builtin_fabsf(x));   // in (0,1], no overflow
  float r = (1.0f - e) * __builtin_amdgcn_rcpf(1.0f + e);
  return __builtin_copysignf(r, x);
}

// Per-group barrier: only the 8 blocks of one recurrent chain participate.
__device__ __forceinline__ void groupBarrier(unsigned* cnt, unsigned* rel, unsigned phase) {
  __syncthreads();
  if (threadIdx.x == 0) {
    __threadfence();
    unsigned prev = __hip_atomic_fetch_add(cnt, 1u, __ATOMIC_ACQ_REL,
                                           __HIP_MEMORY_SCOPE_AGENT);
    if (prev == NBLK_G - 1u) {
      __hip_atomic_store(cnt, 0u, __ATOMIC_RELAXED, __HIP_MEMORY_SCOPE_AGENT);
      __hip_atomic_fetch_add(rel, 1u, __ATOMIC_RELEASE, __HIP_MEMORY_SCOPE_AGENT);
    } else {
      while (__hip_atomic_load(rel, __ATOMIC_ACQUIRE, __HIP_MEMORY_SCOPE_AGENT) < phase) {
        __builtin_amdgcn_s_sleep(1);
      }
    }
  }
  __syncthreads();
}

__global__ void k_zero_sync(unsigned* p) {
  // 4 groups x 64-uint (256 B) slots
  if (threadIdx.x < NGROUP * 64) p[threadIdx.x] = 0u;
}

__global__ void k_f2bf(const float* __restrict__ s, unsigned short* __restrict__ d, int n) {
  int i = blockIdx.x * blockDim.x + threadIdx.x;
  if (i < n) d[i] = f2bf_rne(s[i]);
}

// mx[s][b][i] = bf16( inp[s][b][i] * h_mask[b][i] ), 4-wide
__global__ void k_maskin(const float4* __restrict__ inp4, const float* __restrict__ hmask,
                         v4us* __restrict__ mx4, int total4) {
  int i = blockIdx.x * blockDim.x + threadIdx.x;
  if (i < total4) {
    float4 v = inp4[i];
    int base = (i * 4) & (B_DIM * IN_DIM - 1);
    const float4 m = *(const float4*)(hmask + base);
    v4us r;
    r[0] = f2bf_rne(v.x * m.x);
    r[1] = f2bf_rne(v.y * m.y);
    r[2] = f2bf_rne(v.z * m.z);
    r[3] = f2bf_rne(v.w * m.w);
    mx4[i] = r;
  }
}

__global__ __launch_bounds__(TPB) void k_rhn(
    const unsigned short* __restrict__ mxbf,   // [S][B][IN] bf16
    const unsigned short* __restrict__ Wihb,   // [HID][IN]  bf16
    const unsigned short* __restrict__ Witb,
    const unsigned short* __restrict__ Wicb,
    const unsigned short* __restrict__ Whb,    // [DEPTH][HID][HID] bf16
    const unsigned short* __restrict__ Wtb,
    unsigned short* __restrict__ smbf,         // 2 * [B][HID] bf16 (ping-pong)
    const float* __restrict__ bih, const float* __restrict__ bitp,
    const float* __restrict__ bic, const float* __restrict__ bh,
    const float* __restrict__ bt,
    const float* __restrict__ hidden,          // [B][HID]
    const float* __restrict__ shmask,          // [B][HID]
    float* __restrict__ out,                   // outs [S][B][HID] ++ last [B][HID]
    unsigned* __restrict__ sync) {
  const int tid   = threadIdx.x;
  const int wave  = (blockIdx.x * TPB + tid) >> 5;  // 0..127
  const int lane  = tid & 31;
  const int half  = lane >> 4;
  const int lr    = lane & 15;
  const int tileB = wave >> 5;                      // 0..3 == blockIdx.x >> 3
  const int tileN = wave & 31;                      // 0..31
  const int b0 = tileB * 16, n0 = tileN * 16;
  const int kA = 8 * half;    // A-fragment K offset
  const int kB = 8 * half;    // B-fragment K offset
  const int ncol = n0 + lr;   // D-tile column for this lane

  // Sync slot for this chain (rows b0..b0+15 never interact with other chains)
  unsigned* gsync = sync + (blockIdx.x >> 3) * 64;

  // Per-lane constants + state in registers: element e -> (b0+e+8*half, ncol)
  float mask_e[8];
  v8f s_reg;
#pragma unroll
  for (int e = 0; e < 8; ++e) {
    int b = b0 + e + 8 * half;
    mask_e[e] = shmask[b * HID_DIM + ncol];
    s_reg[e]  = hidden[b * HID_DIM + ncol];
  }

  // Loop-invariant biases for this lane's column: hoist off the critical path.
  float bhv_l[DEPTH], btv_l[DEPTH];
#pragma unroll
  for (int l = 0; l < DEPTH; ++l) {
    bhv_l[l] = bh[l * HID_DIM + ncol];
    btv_l[l] = bt[l * HID_DIM + ncol];
  }
  const float bihv = bih[ncol], bitv = bitp[ncol], bicv = bic[ncol];

  // Initial masked state into ping-pong buffer 0 (read by stage 0)
#pragma unroll
  for (int e = 0; e < 8; ++e) {
    int b = b0 + e + 8 * half;
    smbf[b * HID_DIM + ncol] = f2bf_rne(mask_e[e] * s_reg[e]);
  }

  unsigned phase = 0;
  unsigned st = 0;
  for (int s = 0; s < S_LEN; ++s) {
#pragma unroll 1
    for (int l = 0; l < DEPTH; ++l, ++st) {
      groupBarrier(&gsync[0], &gsync[1], ++phase);

      const unsigned short* A    = smbf + (st & 1u) * (B_DIM * HID_DIM);
      const unsigned short* arow = A + (b0 + lr) * HID_DIM;
      const unsigned short* whr  = Whb + (size_t)(l * HID_DIM + n0 + lr) * HID_DIM;
      const unsigned short* wtr  = Wtb + (size_t)(l * HID_DIM + n0 + lr) * HID_DIM;

      v8f acc_h = {0.f, 0.f, 0.f, 0.f, 0.f, 0.f, 0.f, 0.f};
      v8f acc_t = {0.f, 0.f, 0.f, 0.f, 0.f, 0.f, 0.f, 0.f};
#pragma unroll 8
      for (int kt = 0; kt < 16; ++kt) {
        v16bf a  = ldA(arow + kt * 32 + kA);
        v16bf wh = ldB(whr + kt * 32 + kB);
        v16bf wt = ldB(wtr + kt * 32 + kB);
        acc_h = wmma_bf16(a, wh, acc_h);
        acc_t = wmma_bf16(a, wt, acc_t);
      }
      const float bhv = bhv_l[l];
      const float btv = btv_l[l];

      if (l == 0) {
        // Fused input projections H/T/C for step s (A = masked input slice)
        v8f aH = {0.f, 0.f, 0.f, 0.f, 0.f, 0.f, 0.f, 0.f};
        v8f aT = {0.f, 0.f, 0.f, 0.f, 0.f, 0.f, 0.f, 0.f};
        v8f aC = {0.f, 0.f, 0.f, 0.f, 0.f, 0.f, 0.f, 0.f};
        const unsigned short* xrow = mxbf + ((size_t)s * B_DIM + b0 + lr) * IN_DIM;
        const unsigned short* wih  = Wihb + (size_t)(n0 + lr) * IN_DIM;
        const unsigned short* wit  = Witb + (size_t)(n0 + lr) * IN_DIM;
        const unsigned short* wic  = Wicb + (size_t)(n0 + lr) * IN_DIM;
#pragma unroll 2
        for (int kt = 0; kt < 16; ++kt) {
          v16bf a2 = ldA(xrow + kt * 32 + kA);
          aH = wmma_bf16(a2, ldB(wih + kt * 32 + kB), aH);
          aT = wmma_bf16(a2, ldB(wit + kt * 32 + kB), aT);
          aC = wmma_bf16(a2, ldB(wic + kt * 32 + kB), aC);
        }
        if (s + 1 < S_LEN) {
          __builtin_prefetch((const void*)(mxbf + ((size_t)(s + 1) * B_DIM + b0 + lr) * IN_DIM),
                             0, 1);
        }
#pragma unroll
        for (int e = 0; e < 8; ++e) {
          float lt = acc_t[e] + btv;
          float h = tanh_fast(aH[e] + bihv + acc_h[e] + bhv);
          float t = sigmf(aT[e] + bitv + lt);
          float c = sigmf(aC[e] + bicv + lt);
          s_reg[e] = h * t + s_reg[e] * c;
        }
      } else {
#pragma unroll
        for (int e = 0; e < 8; ++e) {
          float h = tanh_fast(acc_h[e] + bhv);
          float t = sigmf(acc_t[e] + btv);   // c == t when l > 0 (shared lin_t)
          s_reg[e] = h * t + s_reg[e] * t;
        }
      }

      // Publish masked state for next stage into the other ping-pong buffer
      unsigned short* Aout = smbf + ((st + 1u) & 1u) * (B_DIM * HID_DIM);
#pragma unroll
      for (int e = 0; e < 8; ++e) {
        int b = b0 + e + 8 * half;
        Aout[b * HID_DIM + ncol] = f2bf_rne(mask_e[e] * s_reg[e]);
      }
    }

    // outs[s] = state after the 5 depth layers
    float* orow = out + (size_t)s * (B_DIM * HID_DIM);
#pragma unroll
    for (int e = 0; e < 8; ++e) {
      int b = b0 + e + 8 * half;
      orow[b * HID_DIM + ncol] = s_reg[e];
    }
  }

  // last = outs[-1]
  float* lastp = out + (size_t)S_LEN * (B_DIM * HID_DIM);
#pragma unroll
  for (int e = 0; e < 8; ++e) {
    int b = b0 + e + 8 * half;
    lastp[b * HID_DIM + ncol] = s_reg[e];
  }
}

extern "C" void kernel_launch(void* const* d_in, const int* in_sizes, int n_in,
                              void* d_out, int out_size, void* d_ws, size_t ws_size,
                              hipStream_t stream) {
  (void)in_sizes; (void)n_in; (void)out_size; (void)ws_size;
  const float* inp      = (const float*)d_in[0];
  const float* hidden   = (const float*)d_in[1];
  const float* h_mask   = (const float*)d_in[2];
  const float* s_h_mask = (const float*)d_in[3];
  const float* Wih = (const float*)d_in[4];
  const float* bih = (const float*)d_in[5];
  const float* Wit = (const float*)d_in[6];
  const float* bit = (const float*)d_in[7];
  const float* Wic = (const float*)d_in[8];
  const float* bic = (const float*)d_in[9];
  const float* Wh  = (const float*)d_in[10];
  const float* bh  = (const float*)d_in[11];
  const float* Wt  = (const float*)d_in[12];
  const float* bt  = (const float*)d_in[13];

  char* ws = (char*)d_ws;
  size_t off = 0;
  auto alloc = [&](size_t bytes) -> void* {
    void* p = ws + off;
    off = (off + bytes + 255) & ~(size_t)255;
    return p;
  };
  unsigned short* mxbf = (unsigned short*)alloc((size_t)S_LEN * B_DIM * IN_DIM * 2);
  unsigned short* Wihb = (unsigned short*)alloc((size_t)HID_DIM * IN_DIM * 2);
  unsigned short* Witb = (unsigned short*)alloc((size_t)HID_DIM * IN_DIM * 2);
  unsigned short* Wicb = (unsigned short*)alloc((size_t)HID_DIM * IN_DIM * 2);
  unsigned short* Whb  = (unsigned short*)alloc((size_t)DEPTH * HID_DIM * HID_DIM * 2);
  unsigned short* Wtb  = (unsigned short*)alloc((size_t)DEPTH * HID_DIM * HID_DIM * 2);
  unsigned short* smbf = (unsigned short*)alloc((size_t)2 * B_DIM * HID_DIM * 2);
  unsigned* sync       = (unsigned*)alloc((size_t)NGROUP * 64 * sizeof(unsigned));

  k_zero_sync<<<1, 256, 0, stream>>>(sync);

  const int nw = HID_DIM * IN_DIM;
  k_f2bf<<<(nw + 255) / 256, 256, 0, stream>>>(Wih, Wihb, nw);
  k_f2bf<<<(nw + 255) / 256, 256, 0, stream>>>(Wit, Witb, nw);
  k_f2bf<<<(nw + 255) / 256, 256, 0, stream>>>(Wic, Wicb, nw);
  const int nwh = DEPTH * HID_DIM * HID_DIM;
  k_f2bf<<<(nwh + 255) / 256, 256, 0, stream>>>(Wh, Whb, nwh);
  k_f2bf<<<(nwh + 255) / 256, 256, 0, stream>>>(Wt, Wtb, nwh);

  const int total4 = (S_LEN * B_DIM * IN_DIM) / 4;
  k_maskin<<<(total4 + 255) / 256, 256, 0, stream>>>((const float4*)inp, h_mask,
                                                     (v4us*)mxbf, total4);

  k_rhn<<<NBLK, TPB, 0, stream>>>(mxbf, Wihb, Witb, Wicb, Whb, Wtb, smbf,
                                  bih, bit, bic, bh, bt, hidden, s_h_mask,
                                  (float*)d_out, sync);
}
